// DeepSeekMoE_53137335386279
// MI455X (gfx1250) — matrix operand order repaired and checked by reading, not compiled, not run on hardware
//
#include <hip/hip_runtime.h>
#include <hip/hip_bf16.h>
#include <math.h>

#define DEV __device__ __forceinline__

typedef __attribute__((ext_vector_type(16))) __bf16 v16bf;
typedef __attribute__((ext_vector_type(8)))  float  v8f;

static constexpr int Tn   = 4096;          // B*S tokens
static constexpr int Dn   = 2048;          // model dim
static constexpr int Hn   = 1408;          // hidden dim
static constexpr int En   = 16;            // routed experts
static constexpr int NSn  = 2;             // shared experts
static constexpr int RMAX = 2 * Tn + En * 128;  // 10240 padded routed rows max
static constexpr int RT_MAX = RMAX / 128;       // 80 row tiles
static constexpr int BM = 128, BN = 128, BK = 64;
static constexpr int LDA = BK + 8;         // 72 halves (16B-divisible row stride)
static constexpr int LDB = BK + 8;

union Frag16 { v16bf v; unsigned short u[16]; };

DEV unsigned short f2bf(float f) {
  unsigned int u = __float_as_uint(f);
  u += 0x7fffu + ((u >> 16) & 1u);           // round-to-nearest-even
  return (unsigned short)(u >> 16);
}
DEV float bf2f(unsigned int hw) { return __uint_as_float(hw << 16); }
DEV float silu(float v) { return v / (1.f + __expf(-v)); }

// ---------------- WMMA fragment loads (bf16, 16x16x32 layouts) ----------------
// A tile in LDS: [BM][LDA] halves, K-contiguous.  lane m (0..15), group g:
// halves 0..7 -> K = 8g+i ; halves 8..15 -> K = 16+8g+i
DEV v16bf load_a_frag(const unsigned short* As, int row0, int k0) {
  const int lane = threadIdx.x & 31;
  const int m = lane & 15, g = lane >> 4;
  const unsigned short* p = As + (row0 + m) * LDA + k0 + 8 * g;
  Frag16 f;
#pragma unroll
  for (int i = 0; i < 8; ++i) { f.u[i] = p[i]; f.u[8 + i] = p[16 + i]; }
  return f.v;
}
// B tile in LDS transposed: [BN][LDB] halves.  lane n (0..15), group g:
// halves 0..7 -> K = 8g+i ; halves 8..15 -> K = 16+8g+i
DEV v16bf load_b_frag(const unsigned short* Bs, int col0, int k0) {
  const int lane = threadIdx.x & 31;
  const int n = lane & 15, g = lane >> 4;
  const unsigned short* p = Bs + (col0 + n) * LDB + k0 + 8 * g;
  Frag16 f;
#pragma unroll
  for (int i = 0; i < 8; ++i) { f.u[i] = p[i]; f.u[8 + i] = p[16 + i]; }
  return f.v;
}

DEV void mma_tile(const unsigned short* As, const unsigned short* Bs,
                  int wm, int wn, v8f (&acc)[2][4]) {
#pragma unroll
  for (int ks = 0; ks < BK; ks += 32) {
    v16bf a0 = load_a_frag(As, wm * 32, ks);
    v16bf a1 = load_a_frag(As, wm * 32 + 16, ks);
#pragma unroll
    for (int ni = 0; ni < 4; ++ni) {
      v16bf b = load_b_frag(Bs, wn * 64 + ni * 16, ks);
      acc[0][ni] = __builtin_amdgcn_wmma_f32_16x16x32_bf16(
          false, a0, false, b, (short)0, acc[0][ni], false, false);
      acc[1][ni] = __builtin_amdgcn_wmma_f32_16x16x32_bf16(
          false, a1, false, b, (short)0, acc[1][ni], false, false);
    }
  }
}

// ---------------- global -> LDS tile staging ----------------
// A: src is bf16 row-major [*, ld]; copy 128 rows x 64 halves
DEV void load_a_tile(const unsigned short* src, int ld, int row_base, int k0,
                     unsigned short* As) {
  const int tid = threadIdx.x;
  const int row = tid >> 1, seg = (tid & 1) * 32;
  const uint4* g = (const uint4*)(src + (size_t)(row_base + row) * ld + k0 + seg);
  uint4* s = (uint4*)(As + row * LDA + seg);
  s[0] = g[0]; s[1] = g[1]; s[2] = g[2]; s[3] = g[3];
}
DEV void load_a_tile_token(const unsigned short* src, int ld, int tok, int k0,
                           unsigned short* As) {
  const int tid = threadIdx.x;
  const int row = tid >> 1, seg = (tid & 1) * 32;
  uint4* s = (uint4*)(As + row * LDA + seg);
  if (tok >= 0) {
    const uint4* g = (const uint4*)(src + (size_t)tok * ld + k0 + seg);
    s[0] = g[0]; s[1] = g[1]; s[2] = g[2]; s[3] = g[3];
  } else {
    uint4 z = {0u, 0u, 0u, 0u};
    s[0] = z; s[1] = z; s[2] = z; s[3] = z;
  }
}
// B: w is f32 row-major [K, ldg] (pre-offset to k0 row); stage 64x128 transposed bf16.
DEV void load_b_tile(const float* w, int ldg, int n0, unsigned short* Bs) {
  const int tid = threadIdx.x;
  const int kk = (tid >> 3) * 2;       // k pair 0..62
  const int nc = (tid & 7) * 16;       // 16-col chunk
  const float* r0 = w + (size_t)kk * ldg + n0 + nc;
  const float* r1 = r0 + ldg;
#pragma unroll
  for (int j = 0; j < 4; ++j) {
    float4 a = ((const float4*)r0)[j];
    float4 b = ((const float4*)r1)[j];
    unsigned short* base = Bs + (nc + j * 4) * LDB + kk;
    *(unsigned int*)(base)           = (unsigned)f2bf(a.x) | ((unsigned)f2bf(b.x) << 16);
    *(unsigned int*)(base + LDB)     = (unsigned)f2bf(a.y) | ((unsigned)f2bf(b.y) << 16);
    *(unsigned int*)(base + 2 * LDB) = (unsigned)f2bf(a.z) | ((unsigned)f2bf(b.z) << 16);
    *(unsigned int*)(base + 3 * LDB) = (unsigned)f2bf(a.w) | ((unsigned)f2bf(b.w) << 16);
  }
}

DEV void acc_zero(v8f (&acc)[2][4]) {
#pragma unroll
  for (int mi = 0; mi < 2; ++mi)
#pragma unroll
    for (int ni = 0; ni < 4; ++ni)
#pragma unroll
      for (int r = 0; r < 8; ++r) acc[mi][ni][r] = 0.f;
}

// ================= small kernels =================
__global__ void init_kernel(int* tor, int* counts) {
  int i = blockIdx.x * 256 + threadIdx.x;
  if (i < RMAX) tor[i] = -1;
  if (i < En) counts[i] = 0;
}

__global__ void convert_x_kernel(const float* __restrict__ x,
                                 unsigned short* __restrict__ xb) {
  size_t i = ((size_t)blockIdx.x * 256 + threadIdx.x) * 8;
  float4 a = ((const float4*)(x + i))[0];
  float4 b = ((const float4*)(x + i))[1];
  uint4 o;
  o.x = (unsigned)f2bf(a.x) | ((unsigned)f2bf(a.y) << 16);
  o.y = (unsigned)f2bf(a.z) | ((unsigned)f2bf(a.w) << 16);
  o.z = (unsigned)f2bf(b.x) | ((unsigned)f2bf(b.y) << 16);
  o.w = (unsigned)f2bf(b.z) | ((unsigned)f2bf(b.w) << 16);
  *(uint4*)(xb + i) = o;
}

// per-token gating: logits over 16 experts, softmax, top-2, slot assignment
__global__ void __launch_bounds__(256)
gate_kernel(const float* __restrict__ x, const float* __restrict__ gw,
            float* __restrict__ probs, float* __restrict__ logits_out,
            int* __restrict__ counts, int* __restrict__ tke,
            int* __restrict__ tkp, float* __restrict__ tkw) {
  __shared__ float red[En][257];
  const int t = blockIdx.x;
  const int tid = threadIdx.x;
  const float* xr = x + (size_t)t * Dn + tid * 8;
  float4 xa = ((const float4*)xr)[0];
  float4 xb = ((const float4*)xr)[1];
#pragma unroll
  for (int e = 0; e < En; ++e) {
    const float* g = gw + (size_t)e * Dn + tid * 8;
    float4 ga = ((const float4*)g)[0];
    float4 gb = ((const float4*)g)[1];
    red[e][tid] = xa.x * ga.x + xa.y * ga.y + xa.z * ga.z + xa.w * ga.w +
                  xb.x * gb.x + xb.y * gb.y + xb.z * gb.z + xb.w * gb.w;
  }
  for (int s = 128; s > 0; s >>= 1) {
    __syncthreads();
    if (tid < s) {
#pragma unroll
      for (int e = 0; e < En; ++e) red[e][tid] += red[e][tid + s];
    }
  }
  __syncthreads();
  if (tid == 0) {
    float lg[En], pb[En];
    float mx = -1e30f;
    for (int e = 0; e < En; ++e) { lg[e] = red[e][0]; mx = fmaxf(mx, lg[e]); }
    float ssum = 0.f;
    for (int e = 0; e < En; ++e) { pb[e] = __expf(lg[e] - mx); ssum += pb[e]; }
    float inv = 1.f / ssum;
    for (int e = 0; e < En; ++e) {
      pb[e] *= inv;
      probs[(size_t)t * En + e] = pb[e];
      logits_out[(size_t)t * En + e] = lg[e];
    }
    int i0 = 0;
    for (int e = 1; e < En; ++e) if (pb[e] > pb[i0]) i0 = e;
    int i1 = (i0 == 0) ? 1 : 0;
    for (int e = 0; e < En; ++e) if (e != i0 && pb[e] > pb[i1]) i1 = e;
    float w0 = pb[i0], w1 = pb[i1];
    float wn = 1.f / (w0 + w1);
    w0 *= wn; w1 *= wn;
    int p0 = atomicAdd(&counts[i0], 1);
    int p1 = atomicAdd(&counts[i1], 1);
    tke[t * 2] = i0; tke[t * 2 + 1] = i1;
    tkp[t * 2] = p0; tkp[t * 2 + 1] = p1;
    tkw[t * 2] = w0; tkw[t * 2 + 1] = w1;
  }
}

// aux loss + padded prefix offsets (pad each expert range to 128 rows)
__global__ void __launch_bounds__(256)
scan_aux_kernel(const float* __restrict__ probs, const float* __restrict__ logits,
                const int* __restrict__ counts, int* __restrict__ offpad,
                float* __restrict__ aux_out) {
  __shared__ float red[256];
  __shared__ float sumP[En], sumL[En];
  const int tid = threadIdx.x;
  float pp[En], ll[En];
  for (int e = 0; e < En; ++e) { pp[e] = 0.f; ll[e] = 0.f; }
  for (int t = tid; t < Tn; t += 256) {
#pragma unroll
    for (int e = 0; e < En; ++e) {
      pp[e] += probs[(size_t)t * En + e];
      ll[e] += logits[(size_t)t * En + e];
    }
  }
  for (int e = 0; e < En; ++e) {
    red[tid] = pp[e];
    for (int s = 128; s > 0; s >>= 1) { __syncthreads(); if (tid < s) red[tid] += red[tid + s]; }
    __syncthreads();
    if (tid == 0) sumP[e] = red[0];
    __syncthreads();
    red[tid] = ll[e];
    for (int s = 128; s > 0; s >>= 1) { __syncthreads(); if (tid < s) red[tid] += red[tid + s]; }
    __syncthreads();
    if (tid == 0) sumL[e] = red[0];
    __syncthreads();
  }
  if (tid == 0) {
    const float invT = 1.f / (float)Tn;
    float aux = 0.f;
    for (int e = 0; e < En; ++e) aux += (sumP[e] * invT) * (sumL[e] * invT);
    *aux_out = aux * (float)En;
    int off = 0;
    for (int e = 0; e < En; ++e) {
      offpad[e] = off;
      off += ((counts[e] + 127) / 128) * 128;
    }
    offpad[En] = off;
  }
}

__global__ void scatter_kernel(const int* __restrict__ tke, const int* __restrict__ tkp,
                               const float* __restrict__ tkw, const int* __restrict__ offpad,
                               int* __restrict__ tor, float* __restrict__ roww,
                               int* __restrict__ tkr) {
  int i = blockIdx.x * 256 + threadIdx.x;
  if (i >= Tn * 2) return;
  int t = i >> 1;
  int e = tke[i];
  int row = offpad[e] + tkp[i];
  tor[row] = t;
  roww[row] = tkw[i];
  tkr[i] = row;
}

// ================= GEMM kernels =================
// shared experts GEMM1: Hs[t, s*H + n] = silu(X @ shared_w1[s])
__global__ void __launch_bounds__(256)
gemm_shared1(const unsigned short* __restrict__ Xb, const float* __restrict__ sw1,
             unsigned short* __restrict__ Hs) {
  __shared__ unsigned short As[BM * LDA];
  __shared__ unsigned short Bs[BN * LDB];
  const int row_base = blockIdx.x * BM, n0 = blockIdx.y * BN, s = blockIdx.z;
  const float* w = sw1 + (size_t)s * Dn * Hn;
  const int wid = threadIdx.x >> 5, wm = wid & 3, wn = wid >> 2;
  v8f acc[2][4];
  acc_zero(acc);
  for (int k0 = 0; k0 < Dn; k0 += BK) {
    __syncthreads();
    load_a_tile(Xb, Dn, row_base, k0, As);
    load_b_tile(w + (size_t)k0 * Hn, Hn, n0, Bs);
    __syncthreads();
    mma_tile(As, Bs, wm, wn, acc);
  }
  const int lane = threadIdx.x & 31;
  const int mlo = (lane >> 4) * 8, nn = lane & 15;
#pragma unroll
  for (int mi = 0; mi < 2; ++mi)
#pragma unroll
    for (int ni = 0; ni < 4; ++ni)
#pragma unroll
      for (int r = 0; r < 8; ++r) {
        int m = row_base + wm * 32 + mi * 16 + mlo + r;
        int n = n0 + wn * 64 + ni * 16 + nn;
        Hs[(size_t)m * (NSn * Hn) + s * Hn + n] = f2bf(silu(acc[mi][ni][r]));
      }
}

// shared experts GEMM2: out[t, n] = Hs[t, :] @ shared_w2_flat  (K = NS*H = 2816)
__global__ void __launch_bounds__(256)
gemm_shared2(const unsigned short* __restrict__ Hs, const float* __restrict__ sw2,
             float* __restrict__ out) {
  __shared__ unsigned short As[BM * LDA];
  __shared__ unsigned short Bs[BN * LDB];
  const int row_base = blockIdx.x * BM, n0 = blockIdx.y * BN;
  const int wid = threadIdx.x >> 5, wm = wid & 3, wn = wid >> 2;
  const int K = NSn * Hn;
  v8f acc[2][4];
  acc_zero(acc);
  for (int k0 = 0; k0 < K; k0 += BK) {
    __syncthreads();
    load_a_tile(Hs, K, row_base, k0, As);
    load_b_tile(sw2 + (size_t)k0 * Dn, Dn, n0, Bs);
    __syncthreads();
    mma_tile(As, Bs, wm, wn, acc);
  }
  const int lane = threadIdx.x & 31;
  const int mlo = (lane >> 4) * 8, nn = lane & 15;
#pragma unroll
  for (int mi = 0; mi < 2; ++mi)
#pragma unroll
    for (int ni = 0; ni < 4; ++ni)
#pragma unroll
      for (int r = 0; r < 8; ++r) {
        int m = row_base + wm * 32 + mi * 16 + mlo + r;
        int n = n0 + wn * 64 + ni * 16 + nn;
        out[(size_t)m * Dn + n] = acc[mi][ni][r];
      }
}

// routed GEMM1 (gathered): Hb[row, n] = combine_w[row] * silu(X[token(row)] @ w1[e])
__global__ void __launch_bounds__(256)
gemm_routed1(const unsigned short* __restrict__ Xb, const float* __restrict__ w1,
             const int* __restrict__ tor, const float* __restrict__ roww,
             const int* __restrict__ offpad, unsigned short* __restrict__ Hb) {
  __shared__ unsigned short As[BM * LDA];
  __shared__ unsigned short Bs[BN * LDB];
  const int row_base = blockIdx.x * BM;
  if (row_base >= offpad[En]) return;               // uniform exit
  int e = 0;
  while (e < En - 1 && offpad[e + 1] <= row_base) ++e;
  const float* w = w1 + (size_t)e * Dn * Hn;
  const int n0 = blockIdx.y * BN;
  const int wid = threadIdx.x >> 5, wm = wid & 3, wn = wid >> 2;
  const int tok = tor[row_base + (threadIdx.x >> 1)];  // gather index, K-invariant
  v8f acc[2][4];
  acc_zero(acc);
  for (int k0 = 0; k0 < Dn; k0 += BK) {
    __syncthreads();
    load_a_tile_token(Xb, Dn, tok, k0, As);
    load_b_tile(w + (size_t)k0 * Hn, Hn, n0, Bs);
    __syncthreads();
    mma_tile(As, Bs, wm, wn, acc);
  }
  const int lane = threadIdx.x & 31;
  const int mlo = (lane >> 4) * 8, nn = lane & 15;
#pragma unroll
  for (int mi = 0; mi < 2; ++mi)
#pragma unroll
    for (int ni = 0; ni < 4; ++ni)
#pragma unroll
      for (int r = 0; r < 8; ++r) {
        int grow = row_base + wm * 32 + mi * 16 + mlo + r;
        if (tor[grow] >= 0) {
          int n = n0 + wn * 64 + ni * 16 + nn;
          float sv = roww[grow] * silu(acc[mi][ni][r]);
          Hb[(size_t)grow * Hn + n] = f2bf(sv);
        }
      }
}

// routed GEMM2: Yb[row, n] = Hb[row, :] @ w2[e]
__global__ void __launch_bounds__(256)
gemm_routed2(const unsigned short* __restrict__ Hb, const float* __restrict__ w2,
             const int* __restrict__ offpad, unsigned short* __restrict__ Yb) {
  __shared__ unsigned short As[BM * LDA];
  __shared__ unsigned short Bs[BN * LDB];
  const int row_base = blockIdx.x * BM;
  if (row_base >= offpad[En]) return;
  int e = 0;
  while (e < En - 1 && offpad[e + 1] <= row_base) ++e;
  const float* w = w2 + (size_t)e * Hn * Dn;
  const int n0 = blockIdx.y * BN;
  const int wid = threadIdx.x >> 5, wm = wid & 3, wn = wid >> 2;
  v8f acc[2][4];
  acc_zero(acc);
  for (int k0 = 0; k0 < Hn; k0 += BK) {
    __syncthreads();
    load_a_tile(Hb, Hn, row_base, k0, As);
    load_b_tile(w + (size_t)k0 * Dn, Dn, n0, Bs);
    __syncthreads();
    mma_tile(As, Bs, wm, wn, acc);
  }
  const int lane = threadIdx.x & 31;
  const int mlo = (lane >> 4) * 8, nn = lane & 15;
#pragma unroll
  for (int mi = 0; mi < 2; ++mi)
#pragma unroll
    for (int ni = 0; ni < 4; ++ni)
#pragma unroll
      for (int r = 0; r < 8; ++r) {
        int grow = row_base + wm * 32 + mi * 16 + mlo + r;
        int n = n0 + wn * 64 + ni * 16 + nn;
        Yb[(size_t)grow * Dn + n] = f2bf(acc[mi][ni][r]);
      }
}

// out[t, :] += Y[row0(t), :] + Y[row1(t), :]
__global__ void final_add_kernel(float* __restrict__ out, const unsigned short* __restrict__ Yb,
                                 const int* __restrict__ tkr) {
  const int t = blockIdx.x;
  const int c = blockIdx.y * 1024 + threadIdx.x * 4;
  const int r0 = tkr[t * 2], r1 = tkr[t * 2 + 1];
  float4 o = *(float4*)(out + (size_t)t * Dn + c);
  uint2 y0 = *(const uint2*)(Yb + (size_t)r0 * Dn + c);
  uint2 y1 = *(const uint2*)(Yb + (size_t)r1 * Dn + c);
  o.x += bf2f(y0.x & 0xffffu) + bf2f(y1.x & 0xffffu);
  o.y += bf2f(y0.x >> 16)     + bf2f(y1.x >> 16);
  o.z += bf2f(y0.y & 0xffffu) + bf2f(y1.y & 0xffffu);
  o.w += bf2f(y0.y >> 16)     + bf2f(y1.y >> 16);
  *(float4*)(out + (size_t)t * Dn + c) = o;
}

extern "C" void kernel_launch(void* const* d_in, const int* in_sizes, int n_in,
                              void* d_out, int out_size, void* d_ws, size_t ws_size,
                              hipStream_t stream) {
  (void)in_sizes; (void)n_in; (void)out_size; (void)ws_size;
  const float* x   = (const float*)d_in[0];
  const float* gw  = (const float*)d_in[1];
  const float* sw1 = (const float*)d_in[2];
  const float* sw2 = (const float*)d_in[3];
  const float* w1  = (const float*)d_in[4];
  const float* w2  = (const float*)d_in[5];
  float* out = (float*)d_out;

  char* p = (char*)d_ws;
  auto alloc = [&](size_t bytes) -> char* {
    char* r = p;
    p += (bytes + 255) & ~(size_t)255;
    return r;
  };
  unsigned short* Xb = (unsigned short*)alloc((size_t)Tn * Dn * 2);        // x in bf16
  unsigned short* Hs = (unsigned short*)alloc((size_t)Tn * NSn * Hn * 2);  // shared hidden
  unsigned short* Hb = (unsigned short*)alloc((size_t)RMAX * Hn * 2);      // routed hidden
  unsigned short* Yb = (unsigned short*)alloc((size_t)RMAX * Dn * 2);      // routed out rows
  float* probs   = (float*)alloc((size_t)Tn * En * 4);
  float* logitsb = (float*)alloc((size_t)Tn * En * 4);
  int*   counts  = (int*)alloc(En * 4);
  int*   offpad  = (int*)alloc((En + 1) * 4);
  int*   tor     = (int*)alloc(RMAX * 4);        // row -> token (-1 = pad)
  float* roww    = (float*)alloc(RMAX * 4);      // row -> combine weight
  int*   tke     = (int*)alloc(Tn * 2 * 4);
  int*   tkp     = (int*)alloc(Tn * 2 * 4);
  float* tkw     = (float*)alloc(Tn * 2 * 4);
  int*   tkr     = (int*)alloc(Tn * 2 * 4);      // token slot -> row

  init_kernel<<<(RMAX + 255) / 256, 256, 0, stream>>>(tor, counts);
  convert_x_kernel<<<(Tn * Dn) / (256 * 8), 256, 0, stream>>>(x, Xb);
  gate_kernel<<<Tn, 256, 0, stream>>>(x, gw, probs, logitsb, counts, tke, tkp, tkw);
  scan_aux_kernel<<<1, 256, 0, stream>>>(probs, logitsb, counts, offpad,
                                         out + (size_t)Tn * Dn);
  scatter_kernel<<<(Tn * 2) / 256, 256, 0, stream>>>(tke, tkp, tkw, offpad, tor, roww, tkr);
  gemm_shared1<<<dim3(Tn / BM, Hn / BN, NSn), 256, 0, stream>>>(Xb, sw1, Hs);
  gemm_shared2<<<dim3(Tn / BM, Dn / BN), 256, 0, stream>>>(Hs, sw2, out);
  gemm_routed1<<<dim3(RT_MAX, Hn / BN), 256, 0, stream>>>(Xb, w1, tor, roww, offpad, Hb);
  gemm_routed2<<<dim3(RT_MAX, Dn / BN), 256, 0, stream>>>(Hb, w2, offpad, Yb);
  final_add_kernel<<<dim3(Tn, Dn / 1024), 256, 0, stream>>>(out, Yb, tkr);
}
